// WSITokenizer_24189255811387
// MI455X (gfx1250) — compile-verified, matrix operand order checked by reading, not run on hardware
//
#include <hip/hip_runtime.h>

#define EMBED   512
#define NPROTO  32
#define INDIM   1536
#define BATCH   4
#define NTOK    32768
#define EPSV    1e-5f

#define TILE    32            // kv rows per inner iteration
#define RPB     512           // kv rows per block
#define NB      (NTOK / RPB)  // 64 blocks per batch
#define NBLK    (BATCH * NB)  // 256 blocks total
#define SCALEV  0.04419417382415922f  // 1/sqrt(512)

// padded LDS strides (all 16B-aligned fragment chunks, spread banks)
#define RM_STRIDE 1544        // row-major kvn tile: [TILE][RM_STRIDE] f16
#define CM_STRIDE 40          // col-major kvn tile: [INDIM][CM_STRIDE] f16
#define WB_STRIDE 40          // softmax weights:    [NPROTO][WB_STRIDE] f16

typedef __attribute__((ext_vector_type(16))) _Float16 v16h;
typedef __attribute__((ext_vector_type(8)))  _Float16 v8h;
typedef __attribute__((ext_vector_type(4)))  _Float16 v4h;
typedef __attribute__((ext_vector_type(8)))  float    v8f;

static __device__ __forceinline__ v16h cat16(v8h lo, v8h hi) {
    v16h r;
#pragma unroll
    for (int i = 0; i < 8; ++i) { r[i] = lo[i]; r[i + 8] = hi[i]; }
    return r;
}

// ---------------------------------------------------------------------------
// K0a: Q = layernorm(prototypes); qkb[p] = dot(Q[p], k_b).  One block.
// ---------------------------------------------------------------------------
__global__ __launch_bounds__(512)
void wsi_prepQ(const float* __restrict__ proto, const float* __restrict__ qn_g,
               const float* __restrict__ qn_b, const float* __restrict__ k_b,
               float* __restrict__ Qf, float* __restrict__ qkb)
{
    const int tid = threadIdx.x, wid = tid >> 5, lane = tid & 31;
#pragma unroll 1
    for (int half = 0; half < 2; ++half) {
        const int row = half * 16 + wid;
        const float4* src = (const float4*)(proto + (size_t)row * EMBED);
        float4 v[4];
        float s = 0.f, s2 = 0.f;
#pragma unroll
        for (int i = 0; i < 4; ++i) {
            v[i] = src[i * 32 + lane];
            s  += v[i].x + v[i].y + v[i].z + v[i].w;
            s2 += v[i].x * v[i].x + v[i].y * v[i].y + v[i].z * v[i].z + v[i].w * v[i].w;
        }
#pragma unroll
        for (int off = 16; off > 0; off >>= 1) {
            s += __shfl_xor(s, off, 32);  s2 += __shfl_xor(s2, off, 32);
        }
        const float mu = s * (1.f / EMBED);
        const float rstd = rsqrtf(s2 * (1.f / EMBED) - mu * mu + EPSV);
        float qd = 0.f;
#pragma unroll
        for (int i = 0; i < 4; ++i) {
            const float4 g  = ((const float4*)qn_g)[i * 32 + lane];
            const float4 bb = ((const float4*)qn_b)[i * 32 + lane];
            const float4 kb = ((const float4*)k_b)[i * 32 + lane];
            float4 o;
            o.x = (v[i].x - mu) * rstd * g.x + bb.x;
            o.y = (v[i].y - mu) * rstd * g.y + bb.y;
            o.z = (v[i].z - mu) * rstd * g.z + bb.z;
            o.w = (v[i].w - mu) * rstd * g.w + bb.w;
            ((float4*)(Qf + (size_t)row * EMBED))[i * 32 + lane] = o;
            qd += o.x * kb.x + o.y * kb.y + o.z * kb.z + o.w * kb.w;
        }
#pragma unroll
        for (int off = 16; off > 0; off >>= 1) qd += __shfl_xor(qd, off, 32);
        if (lane == 0) qkb[row] = qd;
    }
}

// ---------------------------------------------------------------------------
// K0b: Qk[p,d] = sum_e Q[p,e] * k_w[e,d]  -> f16   (32x1536)
// ---------------------------------------------------------------------------
__global__ __launch_bounds__(512)
void wsi_prepQK(const float* __restrict__ Qf, const float* __restrict__ k_w,
                _Float16* __restrict__ qkh)
{
    const int id = blockIdx.x * 512 + threadIdx.x;     // 49152 threads
    const int p = id / INDIM;
    const int d = id - p * INDIM;
    const float* qrow = Qf + (size_t)p * EMBED;
    float s = 0.f;
    for (int e = 0; e < EMBED; ++e)
        s += qrow[e] * k_w[(size_t)e * INDIM + d];     // coalesced over d
    qkh[id] = (_Float16)s;
}

// ---------------------------------------------------------------------------
// Main fused kernel: layernorm(kv) -> scores (WMMA) -> exp -> Z-accumulate
// (WMMA).  One pass over kv; per-block partial (Z, l) to workspace.
// ---------------------------------------------------------------------------
__global__ __launch_bounds__(512, 1)
void wsi_main(const float* __restrict__ kv, const unsigned char* __restrict__ mask,
              const float* __restrict__ kvn_g, const float* __restrict__ kvn_b,
              const _Float16* __restrict__ qkh, const float* __restrict__ qkb,
              float* __restrict__ Zpart, float* __restrict__ lpart)
{
    extern __shared__ char smem[];
    _Float16* rm = (_Float16*)smem;                      // [TILE][RM_STRIDE]
    _Float16* cm = rm + (size_t)TILE * RM_STRIDE;        // [INDIM][CM_STRIDE]
    _Float16* wb = cm + (size_t)INDIM * CM_STRIDE;       // [NPROTO][WB_STRIDE]
    float* smask = (float*)(wb + NPROTO * WB_STRIDE);    // [TILE]
    float* lbuf2 = smask + TILE;                         // [4][16]

    const int blk = blockIdx.x;
    const int b  = blk / NB;
    const int nb = blk % NB;
    const int tid = threadIdx.x;
    const int wid = tid >> 5;
    const int lane = tid & 31;
    const int laneHi = lane >> 4;     // 0 or 1
    const int lane16 = lane & 15;
    const int m = wid & 1;            // scores wave: M-tile (rows 0-15 / 16-31)
    const int q = wid >> 1;           // scores wave: p-tile (p 0-15 / 16-31)

    v8f acc[2][6];
#pragma unroll
    for (int a = 0; a < 2; ++a)
#pragma unroll
        for (int t = 0; t < 6; ++t) acc[a][t] = {};

    float lacc = 0.f;
    const float qkb_l = qkb[(wid < 4) ? (q * 16 + lane16) : 0];

    const float* kvbase = kv + ((size_t)b * NTOK + (size_t)nb * RPB) * INDIM;
    const unsigned char* mbase = mask + (size_t)b * NTOK + (size_t)nb * RPB;

    for (int it = 0; it < RPB / TILE; ++it) {
        __syncthreads();   // previous Z-update done reading LDS
        // ---- stage: layernorm 32 rows -> f16 in LDS (two passes over the
        //      row keeps live registers small so Z accumulators don't spill) ----
#pragma unroll 1
        for (int half = 0; half < 2; ++half) {
            const int row = half * 16 + wid;
            const float4* src = (const float4*)(kvbase + ((size_t)it * TILE + row) * INDIM);
            float s = 0.f, s2 = 0.f;
#pragma unroll 4
            for (int i = 0; i < 12; ++i) {
                const float4 v = src[i * 32 + lane];
                s  += v.x + v.y + v.z + v.w;
                s2 += v.x * v.x + v.y * v.y + v.z * v.z + v.w * v.w;
            }
#pragma unroll
            for (int off = 16; off > 0; off >>= 1) {
                s += __shfl_xor(s, off, 32);  s2 += __shfl_xor(s2, off, 32);
            }
            const float mu = s * (1.f / INDIM);
            const float rstd = rsqrtf(s2 * (1.f / INDIM) - mu * mu + EPSV);
#pragma unroll 2
            for (int i = 0; i < 12; ++i) {
                const int d = (i * 32 + lane) * 4;
                const float4 v  = src[i * 32 + lane];      // re-read, cache-hot
                const float4 g  = ((const float4*)kvn_g)[i * 32 + lane];
                const float4 bb = ((const float4*)kvn_b)[i * 32 + lane];
                const float o0 = (v.x - mu) * rstd * g.x + bb.x;
                const float o1 = (v.y - mu) * rstd * g.y + bb.y;
                const float o2 = (v.z - mu) * rstd * g.z + bb.z;
                const float o3 = (v.w - mu) * rstd * g.w + bb.w;
                v4h h = { (_Float16)o0, (_Float16)o1, (_Float16)o2, (_Float16)o3 };
                *(v4h*)(rm + (size_t)row * RM_STRIDE + d) = h;   // ds_store_b64
                cm[(size_t)(d + 0) * CM_STRIDE + row] = h[0];
                cm[(size_t)(d + 1) * CM_STRIDE + row] = h[1];
                cm[(size_t)(d + 2) * CM_STRIDE + row] = h[2];
                cm[(size_t)(d + 3) * CM_STRIDE + row] = h[3];
            }
        }
        if (tid < TILE) smask[tid] = mbase[it * TILE + tid] ? -1e30f : 0.f;
        __syncthreads();
        // ---- scores: waves 0..3, S(16x16) = kvn(16x1536) x Qk^T ----
        if (wid < 4) {
            v8f c = {};
            const _Float16* aBase = rm + (size_t)(m * 16 + lane16) * RM_STRIDE + laneHi * 8;
            const _Float16* bBase = qkh + (size_t)(q * 16 + lane16) * INDIM + laneHi * 8;
#pragma unroll 4
            for (int k = 0; k < INDIM / 32; ++k) {
                const int kb = k * 32;
                v16h a  = cat16(*(const v8h*)(aBase + kb), *(const v8h*)(aBase + kb + 16));
                v16h bf = cat16(*(const v8h*)(bBase + kb), *(const v8h*)(bBase + kb + 16));
                c = __builtin_amdgcn_wmma_f32_16x16x32_f16(false, a, false, bf,
                                                           (short)0, c, false, false);
            }
            // per-lane p is fixed; rows vary over VGPR index
            const int p = q * 16 + lane16;
#pragma unroll
            for (int r = 0; r < 8; ++r) {
                const int row = m * 16 + r + 8 * laneHi;
                const float sc = (c[r] + qkb_l) * SCALEV + smask[row];
                const float w = __expf(sc);      // masked -> exp(-1e30) == 0
                lacc += w;
                wb[(size_t)p * WB_STRIDE + row] = (_Float16)w;
            }
        }
        __syncthreads();
        // ---- Z update: all 16 waves, Z(32 x 96-slice) += w^T x kvn ----
        {
            v16h afr[2];
#pragma unroll
            for (int qq = 0; qq < 2; ++qq) {
                const _Float16* base = wb + (size_t)(qq * 16 + lane16) * WB_STRIDE + laneHi * 8;
                afr[qq] = cat16(*(const v8h*)base, *(const v8h*)(base + 16));
            }
#pragma unroll
            for (int t = 0; t < 6; ++t) {
                const int d = wid * 96 + t * 16 + lane16;
                const _Float16* base = cm + (size_t)d * CM_STRIDE + laneHi * 8;
                v16h bf = cat16(*(const v8h*)base, *(const v8h*)(base + 16));
                acc[0][t] = __builtin_amdgcn_wmma_f32_16x16x32_f16(false, afr[0], false, bf,
                                                                   (short)0, acc[0][t], false, false);
                acc[1][t] = __builtin_amdgcn_wmma_f32_16x16x32_f16(false, afr[1], false, bf,
                                                                   (short)0, acc[1][t], false, false);
            }
        }
    }
    // ---- epilogue: deterministic l reduction + Z partial store ----
    if (wid < 4) {
        const float lsum = lacc + __shfl_xor(lacc, 16, 32);  // combine row halves
        if (lane < 16) lbuf2[wid * 16 + lane] = lsum;
    }
    float* zp = Zpart + (size_t)blk * NPROTO * INDIM + wid * 96;
#pragma unroll
    for (int qq = 0; qq < 2; ++qq)
#pragma unroll
        for (int t = 0; t < 6; ++t)
#pragma unroll
            for (int r = 0; r < 8; ++r)
                zp[(size_t)(qq * 16 + r + 8 * laneHi) * INDIM + t * 16 + lane16] = acc[qq][t][r];
    __syncthreads();
    if (tid < NPROTO) {
        const int qq = tid >> 4, pl = tid & 15;
        lpart[blk * NPROTO + tid] = lbuf2[(qq * 2 + 0) * 16 + pl] + lbuf2[(qq * 2 + 1) * 16 + pl];
    }
}

// ---------------------------------------------------------------------------
// K3: per (b,p): reduce partials, attended = (Z/L)·Wv^T + v_b,
//     out = attended·Wo^T + out_b, final layernorm.   128 blocks x 512.
// ---------------------------------------------------------------------------
__global__ __launch_bounds__(512)
void wsi_finish(const float* __restrict__ Zpart, const float* __restrict__ lpart,
                const float* __restrict__ v_w, const float* __restrict__ v_b,
                const float* __restrict__ out_w, const float* __restrict__ out_b,
                const float* __restrict__ on_g, const float* __restrict__ on_b,
                float* __restrict__ out)
{
    __shared__ float zrow[INDIM];
    __shared__ float att[EMBED];
    __shared__ float lred[NB];
    __shared__ float redS[16], redS2[16];
    __shared__ float stat[3];   // 1/L, mu, rstd
    const int bp = blockIdx.x, b = bp >> 5, p = bp & 31;
    const int tid = threadIdx.x, wid = tid >> 5, lane = tid & 31;

#pragma unroll
    for (int j = 0; j < 3; ++j) {
        const int d = j * 512 + tid;
        float s = 0.f;
        for (int k = 0; k < NB; ++k)
            s += Zpart[((size_t)(b * NB + k) * NPROTO + p) * INDIM + d];
        zrow[d] = s;
    }
    if (tid < NB) lred[tid] = lpart[(b * NB + tid) * NPROTO + p];
    __syncthreads();
    if (tid == 0) {
        float s = 0.f;
        for (int k = 0; k < NB; ++k) s += lred[k];
        stat[0] = 1.f / s;
    }
    __syncthreads();
    const float linv = stat[0];
    {
        const float* wrow = v_w + (size_t)tid * INDIM;
        float s = 0.f;
        for (int d = 0; d < INDIM; ++d) s += zrow[d] * wrow[d];
        att[tid] = s * linv + v_b[tid];
    }
    __syncthreads();
    float o = out_b[tid];
    {
        const float* orow = out_w + (size_t)tid * EMBED;
        for (int e = 0; e < EMBED; ++e) o += att[e] * orow[e];
    }
    float s = o, s2 = o * o;
#pragma unroll
    for (int off = 16; off > 0; off >>= 1) {
        s += __shfl_xor(s, off, 32);  s2 += __shfl_xor(s2, off, 32);
    }
    if (lane == 0) { redS[wid] = s; redS2[wid] = s2; }
    __syncthreads();
    if (tid == 0) {
        float a = 0.f, qq = 0.f;
        for (int i = 0; i < 16; ++i) { a += redS[i]; qq += redS2[i]; }
        const float mu = a * (1.f / EMBED);
        stat[1] = mu;
        stat[2] = rsqrtf(qq * (1.f / EMBED) - mu * mu + EPSV);
    }
    __syncthreads();
    out[(size_t)bp * EMBED + tid] = (o - stat[1]) * stat[2] * on_g[tid] + on_b[tid];
}

// ---------------------------------------------------------------------------
extern "C" void kernel_launch(void* const* d_in, const int* in_sizes, int n_in,
                              void* d_out, int out_size, void* d_ws, size_t ws_size,
                              hipStream_t stream)
{
    (void)in_sizes; (void)n_in; (void)out_size; (void)ws_size;
    const float* kv            = (const float*)d_in[0];
    const unsigned char* mask  = (const unsigned char*)d_in[1];
    const float* proto         = (const float*)d_in[2];
    const float* k_w           = (const float*)d_in[3];
    const float* k_b           = (const float*)d_in[4];
    const float* v_w           = (const float*)d_in[5];
    const float* v_b           = (const float*)d_in[6];
    const float* kvn_g         = (const float*)d_in[7];
    const float* kvn_b         = (const float*)d_in[8];
    const float* qn_g          = (const float*)d_in[9];
    const float* qn_b          = (const float*)d_in[10];
    const float* on_g          = (const float*)d_in[11];
    const float* on_b          = (const float*)d_in[12];
    const float* out_w         = (const float*)d_in[13];
    const float* out_b         = (const float*)d_in[14];
    float* out = (float*)d_out;

    char* ws = (char*)d_ws;
    float* Qf     = (float*)ws;   ws += (size_t)NPROTO * EMBED * 4;       // 64 KB
    float* qkb    = (float*)ws;   ws += 256;
    _Float16* qkh = (_Float16*)ws; ws += (size_t)NPROTO * INDIM * 2;      // 96 KB
    float* lpart  = (float*)ws;   ws += (size_t)NBLK * NPROTO * 4;        // 32 KB
    float* Zpart  = (float*)ws;   ws += (size_t)NBLK * NPROTO * INDIM * 4; // ~48 MB

    const size_t smemBytes =
        ((size_t)TILE * RM_STRIDE + (size_t)INDIM * CM_STRIDE + (size_t)NPROTO * WB_STRIDE) * 2
        + (TILE + 64) * 4;   // ~224.6 KB of the 320 KB WGP LDS
    hipFuncSetAttribute((const void*)wsi_main,
                        hipFuncAttributeMaxDynamicSharedMemorySize, (int)smemBytes);

    wsi_prepQ<<<1, 512, 0, stream>>>(proto, qn_g, qn_b, k_b, Qf, qkb);
    wsi_prepQK<<<(NPROTO * INDIM) / 512, 512, 0, stream>>>(Qf, k_w, qkh);
    wsi_main<<<NBLK, 512, smemBytes, stream>>>(kv, mask, kvn_g, kvn_b, qkh, qkb, Zpart, lpart);
    wsi_finish<<<BATCH * NPROTO, 512, 0, stream>>>(Zpart, lpart, v_w, v_b,
                                                   out_w, out_b, on_g, on_b, out);
}
